// UNIFORMGTLayer_68453188764120
// MI455X (gfx1250) — compile-verified, matrix operand order checked by reading, not run on hardware
//
#include <hip/hip_runtime.h>
#include <hip/hip_bf16.h>

typedef _Float16 h16;
typedef __attribute__((ext_vector_type(16))) _Float16 v16h;
typedef __attribute__((ext_vector_type(8)))  _Float16 v8h;
typedef __attribute__((ext_vector_type(8)))  float    v8f;

#define DD 128
#define HH 8
#define HDD 16

// ---------- helpers ----------
__device__ __forceinline__ unsigned fkey(float f) {
  unsigned u = __float_as_uint(f);
  return (u & 0x80000000u) ? ~u : (u | 0x80000000u);
}
__device__ __forceinline__ float funkey(unsigned k) {
  return (k & 0x80000000u) ? __uint_as_float(k ^ 0x80000000u)
                           : __uint_as_float(~k);
}

// ---------- elementwise ----------
__global__ void k_cast16(const float* __restrict__ in, h16* __restrict__ out, size_t n) {
  size_t i = (size_t)blockIdx.x * blockDim.x + threadIdx.x;
  size_t st = (size_t)gridDim.x * blockDim.x;
  for (; i < n; i += st) out[i] = (h16)in[i];
}

__global__ void k_fill_f(float* __restrict__ p, float v, size_t n) {
  size_t i = (size_t)blockIdx.x * blockDim.x + threadIdx.x;
  size_t st = (size_t)gridDim.x * blockDim.x;
  for (; i < n; i += st) p[i] = v;
}

// ---------- WMMA GEMM with LDS-staged A tile ----------
//  C[M x Nout] = A[M x Kd](f16, row-major) @ W[Nout x Kd]^T  (+ bias) (+ residual) (+ relu)
//  Block = 256 threads = 8 waves. blockIdx.x -> 16-row M tile; blockIdx.y -> 128-col N slab.
//  A tile (16 x Kd halves) staged once in LDS (padded stride, 4-bank rotation per row),
//  each wave computes one 16x16 tile via v_wmma_f32_16x16x32_f16.
#define LDS_PAD 8
__global__ __launch_bounds__(256)
void k_gemm_wmma(const h16* __restrict__ A, const h16* __restrict__ W,
                 const float* __restrict__ bias, const float* __restrict__ res,
                 float* __restrict__ C, int M, int Kd, int Nout, int relu) {
  __shared__ h16 As[16 * (256 + LDS_PAD)];   // up to Kd=256
  const int ldk  = Kd + LDS_PAD;
  const int tid  = threadIdx.x;
  const int wave = tid >> 5;
  const int lane = tid & 31;
  const int mBase = blockIdx.x * 16;
  const int nBase = blockIdx.y * 128 + wave * 16;

  // Cooperative stage of A tile: 16 rows x Kd halves, 16B chunks.
  const int nchunk = (16 * Kd) >> 11;        // (16*Kd)/(8*256): 1 (Kd=128) or 2 (Kd=256)
  for (int c = 0; c < nchunk; ++c) {
    int idx = (c * 256 + tid) * 8;           // half-index within un-padded tile
    int row = idx / Kd;
    int col = idx - row * Kd;
    *(v8h*)&As[row * ldk + col] =
        *(const v8h*)(A + (size_t)(mBase + row) * Kd + col);
  }
  __syncthreads();

  // A layout (16-bit 16x32): lanes 0-15 row=l16 hold K {0..7,16..23};
  // lanes 16-31 same rows hold K {8..15,24..31}. B mirrored from W rows (B = W^T cols).
  const int hi  = lane >> 4;
  const int l16 = lane & 15;
  const h16* alds = As + l16 * ldk + hi * 8;
  const h16* wrow = W + (size_t)(nBase + l16) * Kd + hi * 8;

  v8f acc = {};
  for (int k0 = 0; k0 < Kd; k0 += 32) {
    __builtin_prefetch(wrow + k0 + 64, 0, 0);
    v8h a0 = *(const v8h*)(alds + k0);
    v8h a1 = *(const v8h*)(alds + k0 + 16);
    v8h b0 = *(const v8h*)(wrow + k0);
    v8h b1 = *(const v8h*)(wrow + k0 + 16);
    v16h a = __builtin_shufflevector(a0, a1, 0,1,2,3,4,5,6,7,8,9,10,11,12,13,14,15);
    v16h b = __builtin_shufflevector(b0, b1, 0,1,2,3,4,5,6,7,8,9,10,11,12,13,14,15);
    acc = __builtin_amdgcn_wmma_f32_16x16x32_f16(
        /*neg_a=*/false, a, /*neg_b=*/false, b,
        /*c_mod=*/(short)0, acc, /*reuse_a=*/false, /*reuse_b=*/false);
  }

  const int col = nBase + l16;
  const float bv = bias ? bias[col] : 0.0f;
#pragma unroll
  for (int r = 0; r < 8; ++r) {
    int row = mBase + hi * 8 + r;            // C/D layout: VGPR r -> M=r / M=8+r
    float v = acc[r] + bv;
    if (relu) v = fmaxf(v, 0.0f);
    if (res) v += res[(size_t)row * Nout + col];
    C[(size_t)row * Nout + col] = v;
  }
}

// ---------- GCN ----------
__global__ void k_deg(const int* __restrict__ ei, float* __restrict__ deg, int E) {
  int i = blockIdx.x * blockDim.x + threadIdx.x;
  if (i < E) atomicAdd(&deg[ei[E + i]], 1.0f);
}
__global__ void k_rsqrt_inplace(float* __restrict__ p, size_t n) {
  size_t i = (size_t)blockIdx.x * blockDim.x + threadIdx.x;
  if (i < n) p[i] = rsqrtf(p[i]);
}
__global__ void k_gcn_init(const float* __restrict__ xw, const float* __restrict__ dinv,
                           const float* __restrict__ bg, float* __restrict__ h2, int N) {
  size_t i = (size_t)blockIdx.x * blockDim.x + threadIdx.x;
  size_t total = (size_t)N * DD;
  if (i < total) {
    int n = (int)(i / DD), d = (int)(i % DD);
    float di = dinv[n];
    h2[i] = di * di * xw[i] + bg[d];         // self-loop term + bias
  }
}
__global__ void k_gcn_agg(const int* __restrict__ ei, const float* __restrict__ xw,
                          const float* __restrict__ dinv, float* __restrict__ h2, int E) {
  size_t t = (size_t)blockIdx.x * blockDim.x + threadIdx.x;
  size_t total = (size_t)E * 32;
  if (t >= total) return;
  int j = (int)(t & 31);
  size_t e = t >> 5;
  int s = ei[e], d = ei[(size_t)E + e];
  float c = dinv[s] * dinv[d];
  const float* sp = xw + (size_t)s * DD + j * 4;
  float* dp = h2 + (size_t)d * DD + j * 4;
#pragma unroll
  for (int k = 0; k < 4; ++k) atomicAdd(&dp[k], sp[k] * c);
}

// ---------- BatchNorm (training, biased stats, eps=1e-5) ----------
__global__ void k_colreduce(const float* __restrict__ in, float* __restrict__ gsum,
                            float* __restrict__ gsq, int N, int C) {
  int c = threadIdx.x;                       // blockDim.x == C
  float s = 0.f, q = 0.f;
  for (int r = blockIdx.x; r < N; r += gridDim.x) {
    float v = in[(size_t)r * C + c];
    s += v; q += v * v;
  }
  atomicAdd(&gsum[c], s);
  atomicAdd(&gsq[c], q);
}
__global__ void k_bnfinish(const float* __restrict__ gsum, const float* __restrict__ gsq,
                           const float* __restrict__ g, const float* __restrict__ b,
                           float* __restrict__ scale, float* __restrict__ shift,
                           int N, int C) {
  int c = threadIdx.x;
  if (c < C) {
    float rn = 1.0f / (float)N;
    float m  = gsum[c] * rn;
    float var = gsq[c] * rn - m * m;
    float sc = g[c] * rsqrtf(var + 1e-5f);
    scale[c] = sc;
    shift[c] = b[c] - m * sc;
  }
}
__global__ void k_bnapply(const float* __restrict__ in, const float* __restrict__ scale,
                          const float* __restrict__ shift, float* __restrict__ out32,
                          h16* __restrict__ out16, size_t n, int C) {
  size_t i = (size_t)blockIdx.x * blockDim.x + threadIdx.x;
  size_t st = (size_t)gridDim.x * blockDim.x;
  for (; i < n; i += st) {
    int c = (int)(i % C);
    float y = in[i] * scale[c] + shift[c];
    if (out32) out32[i] = y;
    if (out16) out16[i] = (h16)y;
  }
}

// ---------- Sparse attention ----------
__global__ void k_init_attn(unsigned* __restrict__ mx, float* __restrict__ ssum, size_t n) {
  size_t i = (size_t)blockIdx.x * blockDim.x + threadIdx.x;
  if (i < n) { mx[i] = 0u; ssum[i] = 1e-16f; }
}
__global__ void k_scores(const int* __restrict__ ei, const float* __restrict__ K,
                         const float* __restrict__ x, float* __restrict__ sc,
                         unsigned* __restrict__ mx, int E) {
  size_t t = (size_t)blockIdx.x * blockDim.x + threadIdx.x;
  size_t total = (size_t)E * HH;
  if (t >= total) return;
  int h = (int)(t % HH);
  size_t e = t / HH;
  int s = ei[e], d = ei[(size_t)E + e];
  const float* kp = K + (size_t)s * DD + h * HDD;
  const float* qp = x + (size_t)d * DD + h * HDD;
  float acc = 0.f;
#pragma unroll
  for (int i = 0; i < HDD; ++i) acc += kp[i] * qp[i];
  acc *= 0.25f;                              // 1/sqrt(16)
  sc[t] = acc;
  atomicMax(&mx[(size_t)d * HH + h], fkey(acc));
}
__global__ void k_expsum(const int* __restrict__ ei, float* __restrict__ sc,
                         const unsigned* __restrict__ mx, float* __restrict__ ssum, int E) {
  size_t t = (size_t)blockIdx.x * blockDim.x + threadIdx.x;
  size_t total = (size_t)E * HH;
  if (t >= total) return;
  int h = (int)(t % HH);
  size_t e = t / HH;
  int d = ei[(size_t)E + e];
  float m = funkey(mx[(size_t)d * HH + h]);
  float ev = expf(sc[t] - m);
  sc[t] = ev;
  atomicAdd(&ssum[(size_t)d * HH + h], ev);
}
__global__ void k_aggmsg(const int* __restrict__ ei, const float* __restrict__ sc,
                         const float* __restrict__ ssum, const float* __restrict__ V,
                         float* __restrict__ agg, int E) {
  size_t t = (size_t)blockIdx.x * blockDim.x + threadIdx.x;
  size_t total = (size_t)E * 32;
  if (t >= total) return;
  int j = (int)(t & 31);
  size_t e = t >> 5;
  int s = ei[e], d = ei[(size_t)E + e];
  int dblk = j * 4;
  int h = dblk >> 4;
  float a = sc[e * HH + h] / ssum[(size_t)d * HH + h];
  const float* vp = V + (size_t)s * DD + dblk;
  float* op = agg + (size_t)d * DD + dblk;
#pragma unroll
  for (int k = 0; k < 4; ++k) atomicAdd(&op[k], vp[k] * a);
}

// ---------- orchestration ----------
extern "C" void kernel_launch(void* const* d_in, const int* in_sizes, int n_in,
                              void* d_out, int out_size, void* d_ws, size_t ws_size,
                              hipStream_t stream) {
  const float* x    = (const float*)d_in[0];
  const int*   ei   = (const int*)d_in[1];
  const float* Wg   = (const float*)d_in[2];
  const float* bg   = (const float*)d_in[3];
  const float* g_kv = (const float*)d_in[4];
  const float* b_kv = (const float*)d_in[5];
  const float* Wk   = (const float*)d_in[6];
  const float* bk   = (const float*)d_in[7];
  const float* Wv   = (const float*)d_in[8];
  const float* bv   = (const float*)d_in[9];
  const float* Wo   = (const float*)d_in[10];
  const float* bo   = (const float*)d_in[11];
  const float* g_at = (const float*)d_in[12];
  const float* b_at = (const float*)d_in[13];
  const float* W1   = (const float*)d_in[14];
  const float* b1   = (const float*)d_in[15];
  const float* W2   = (const float*)d_in[16];
  const float* b2   = (const float*)d_in[17];
  const float* g2   = (const float*)d_in[18];
  const float* b2n  = (const float*)d_in[19];

  const int N = in_sizes[0] / DD;            // 50000
  const int E = in_sizes[1] / 2;             // 600000
  const size_t ND  = (size_t)N * DD;
  const size_t N2D = (size_t)N * 2 * DD;

  char* ws = (char*)d_ws;
  size_t off = 0;
  auto alloc = [&](size_t bytes) -> void* {
    void* p = ws + off;
    off += (bytes + 255) & ~(size_t)255;
    return p;
  };

  h16* Wg16 = (h16*)alloc(DD * DD * 2);
  h16* Wk16 = (h16*)alloc(DD * DD * 2);
  h16* Wv16 = (h16*)alloc(DD * DD * 2);
  h16* Wo16 = (h16*)alloc(DD * DD * 2);
  h16* W116 = (h16*)alloc(2 * DD * DD * 2);
  h16* W216 = (h16*)alloc(2 * DD * DD * 2);
  h16* act16   = (h16*)alloc(ND * 2);        // reused: x16 -> h2_16 -> agg16 -> hattn16
  h16* ffn1_16 = (h16*)alloc(N2D * 2);
  float*    dinv = (float*)alloc((size_t)N * 4);
  unsigned* mx   = (unsigned*)alloc((size_t)N * HH * 4);
  float*    ssum = (float*)alloc((size_t)N * HH * 4);
  float* gsum = (float*)alloc(256 * 4);
  float* gsq  = (float*)alloc(256 * 4);
  float* bsc  = (float*)alloc(256 * 4);
  float* bsh  = (float*)alloc(256 * 4);
  float* esc  = (float*)alloc((size_t)E * HH * 4);
  float* bufA  = (float*)alloc(ND * 4);      // xw
  float* h2    = (float*)alloc(ND * 4);
  float* Kb    = (float*)alloc(ND * 4);      // K -> pre1
  float* Vb    = (float*)alloc(ND * 4);      // V -> pre2
  float* aggV  = (float*)alloc(ND * 4);
  float* hattn = (float*)alloc(ND * 4);
  float* ffn1  = (float*)alloc(N2D * 4);

  auto gs = [](size_t n) { return (unsigned)((n + 255) / 256); };
  const dim3 B256(256);

  // ---- weight + input casts ----
  k_cast16<<<gs(DD * DD), B256, 0, stream>>>(Wg, Wg16, DD * DD);
  k_cast16<<<gs(DD * DD), B256, 0, stream>>>(Wk, Wk16, DD * DD);
  k_cast16<<<gs(DD * DD), B256, 0, stream>>>(Wv, Wv16, DD * DD);
  k_cast16<<<gs(DD * DD), B256, 0, stream>>>(Wo, Wo16, DD * DD);
  k_cast16<<<gs(2 * DD * DD), B256, 0, stream>>>(W1, W116, 2 * DD * DD);
  k_cast16<<<gs(2 * DD * DD), B256, 0, stream>>>(W2, W216, 2 * DD * DD);
  k_cast16<<<gs(ND), B256, 0, stream>>>(x, act16, ND);

  // ---- degrees (self loop => init 1.0) ----
  k_fill_f<<<gs(N), B256, 0, stream>>>(dinv, 1.0f, (size_t)N);
  k_deg<<<gs(E), B256, 0, stream>>>(ei, dinv, E);
  k_rsqrt_inplace<<<gs(N), B256, 0, stream>>>(dinv, (size_t)N);

  // GEMM grids: (M/16 blocks) x (Nout/128 slabs), 256 threads (8 waves) per block
  const dim3 gemmB(256);
  const dim3 gD(N / 16, 1);                  // Nout = 128
  const dim3 g2D(N / 16, 2);                 // Nout = 256

  // ---- GCNConv: xw = x @ Wg^T ; aggregate ; BN(kv) ----
  k_gemm_wmma<<<gD, gemmB, 0, stream>>>(act16, Wg16, nullptr, nullptr, bufA, N, DD, DD, 0);
  k_gcn_init<<<gs(ND), B256, 0, stream>>>(bufA, dinv, bg, h2, N);
  k_gcn_agg<<<gs((size_t)E * 32), B256, 0, stream>>>(ei, bufA, dinv, h2, E);

  k_fill_f<<<1, 256, 0, stream>>>(gsum, 0.0f, 256);
  k_fill_f<<<1, 256, 0, stream>>>(gsq, 0.0f, 256);
  k_colreduce<<<512, DD, 0, stream>>>(h2, gsum, gsq, N, DD);
  k_bnfinish<<<1, DD, 0, stream>>>(gsum, gsq, g_kv, b_kv, bsc, bsh, N, DD);
  k_bnapply<<<gs(ND), B256, 0, stream>>>(h2, bsc, bsh, nullptr, act16, ND, DD);

  // ---- K, V projections ----
  k_gemm_wmma<<<gD, gemmB, 0, stream>>>(act16, Wk16, bk, nullptr, Kb, N, DD, DD, 0);
  k_gemm_wmma<<<gD, gemmB, 0, stream>>>(act16, Wv16, bv, nullptr, Vb, N, DD, DD, 0);

  // ---- edge softmax attention ----
  k_init_attn<<<gs((size_t)N * HH), B256, 0, stream>>>(mx, ssum, (size_t)N * HH);
  k_scores<<<gs((size_t)E * HH), B256, 0, stream>>>(ei, Kb, x, esc, mx, E);
  k_expsum<<<gs((size_t)E * HH), B256, 0, stream>>>(ei, esc, mx, ssum, E);
  k_fill_f<<<gs(ND), B256, 0, stream>>>(aggV, 0.0f, ND);
  k_aggmsg<<<gs((size_t)E * 32), B256, 0, stream>>>(ei, esc, ssum, Vb, aggV, E);

  // ---- output projection (+x residual fused) + BN(attn) ----
  k_cast16<<<gs(ND), B256, 0, stream>>>(aggV, act16, ND);
  k_gemm_wmma<<<gD, gemmB, 0, stream>>>(act16, Wo16, bo, x, Kb, N, DD, DD, 0);  // pre1

  k_fill_f<<<1, 256, 0, stream>>>(gsum, 0.0f, 256);
  k_fill_f<<<1, 256, 0, stream>>>(gsq, 0.0f, 256);
  k_colreduce<<<512, DD, 0, stream>>>(Kb, gsum, gsq, N, DD);
  k_bnfinish<<<1, DD, 0, stream>>>(gsum, gsq, g_at, b_at, bsc, bsh, N, DD);
  k_bnapply<<<gs(ND), B256, 0, stream>>>(Kb, bsc, bsh, hattn, act16, ND, DD);

  // ---- FFN (up+relu, down + hattn residual fused) + BN(2) -> d_out ----
  k_gemm_wmma<<<g2D, gemmB, 0, stream>>>(act16, W116, b1, nullptr, ffn1, N, DD, 2 * DD, 1);
  k_cast16<<<gs(N2D), B256, 0, stream>>>(ffn1, ffn1_16, N2D);
  k_gemm_wmma<<<gD, gemmB, 0, stream>>>(ffn1_16, W216, b2, hattn, Vb, N, 2 * DD, DD, 0); // pre2

  k_fill_f<<<1, 256, 0, stream>>>(gsum, 0.0f, 256);
  k_fill_f<<<1, 256, 0, stream>>>(gsq, 0.0f, 256);
  k_colreduce<<<512, DD, 0, stream>>>(Vb, gsum, gsq, N, DD);
  k_bnfinish<<<1, DD, 0, stream>>>(gsum, gsq, g2, b2n, bsc, bsh, N, DD);
  k_bnapply<<<gs(ND), B256, 0, stream>>>(Vb, bsc, bsh, (float*)d_out, nullptr, ND, DD);
}
